// ApproximateConv2d_66030827209197
// MI455X (gfx1250) — compile-verified, matrix operand order checked by reading, not run on hardware
//
#include <hip/hip_runtime.h>
#include <hip/hip_bf16.h>

typedef __attribute__((ext_vector_type(16))) _Float16 v16h;
typedef __attribute__((ext_vector_type(8)))  _Float16 v8h;
typedef __attribute__((ext_vector_type(8)))  float    v8f;
typedef __attribute__((ext_vector_type(4)))  unsigned v4u;

#define NB 4
#define NC 32
#define HW 56
#define NO 64
#define CKK 288            // 32*3*3
#define NPIX (HW*HW)       // 3136
#define TPI (NPIX/16)      // 196 16-pixel tiles per image
#define WAVES 2
#define THREADS (WAVES*32)
#define XSTR 296           // padded f16 row stride for xu (bank-minimal for b128)
#define NW (NO*CKK)        // 18432 weight elements
#define WCHUNK 2048        // staging chunk: 2048 floats = 8KB

// Packed f16 min without llvm.minnum canonicalization (no v_pk_max_num_f16 x,x).
__device__ static __forceinline__ unsigned pkmin(unsigned a, unsigned b) {
  unsigned r;
  asm("v_pk_min_num_f16 %0, %1, %2" : "=v"(r) : "v"(a), "v"(b));
  return r;
}

__device__ static __forceinline__ v8h min8(v4u xa, v4u wa) {
  v4u m;
  m.x = pkmin(xa.x, wa.x);
  m.y = pkmin(xa.y, wa.y);
  m.z = pkmin(xa.z, wa.z);
  m.w = pkmin(xa.w, wa.w);
  return __builtin_bit_cast(v8h, m);
}

// z[b,o,l] = mu_w * sum_k min(x_unf[k,l], w[o,k])
// mins on VALU (packed f16), reduction over k via v_wmma_f32_16x16x32_f16
// with an all-ones A matrix (D[m,n] = sum_k B[k,n]); C/D accumulate in f32.
__global__ __launch_bounds__(THREADS)
void minplus_conv2d_kernel(const float* __restrict__ x,
                           const float* __restrict__ w,
                           float* __restrict__ out)
{
  __shared__ __align__(16) float    wstg[WCHUNK];          // 8 KB async staging
  __shared__ __align__(16) _Float16 wh[NW];                // 36 KB f16 weights
  __shared__ __align__(16) _Float16 xu[WAVES * 16 * XSTR]; // 18.5 KB unfolded x
  __shared__ float smu;

  const int tid  = threadIdx.x;
  const int wave = tid >> 5;
  const int lane = tid & 31;

  // ---- Stage weights: async global->LDS (CDNA5 path), f32->f16, mean|w| ----
  float musum = 0.f;
  for (int ch = 0; ch < NW / WCHUNK; ++ch) {               // 9 chunks
    for (int it = 0; it < WCHUNK / (THREADS * 4); ++it) {  // 8 b128/thread
      const int e4 = (it * THREADS + tid) * 4;             // float index
      const unsigned lds = (unsigned)(unsigned long long)(uintptr_t)&wstg[e4];
      const int gb = (ch * WCHUNK + e4) * 4;               // byte offset
      asm volatile("global_load_async_to_lds_b128 %0, %1, %2"
                   :: "v"(lds), "v"(gb), "s"(w) : "memory");
    }
    asm volatile("s_wait_asynccnt 0" ::: "memory");
    __syncthreads();
    for (int it = 0; it < WCHUNK / THREADS; ++it) {        // 32 converts/thread
      const int e = it * THREADS + tid;
      const float v = wstg[e];
      musum += __builtin_fabsf(v);
      wh[ch * WCHUNK + e] = (_Float16)v;
    }
    __syncthreads();                                        // before reuse of wstg
  }
  wstg[tid] = musum;
  __syncthreads();
  if (tid == 0) {
    float t = 0.f;
    for (int i = 0; i < THREADS; ++i) t += wstg[i];
    smu = t * (1.0f / (float)NW);
  }

  // ---- Build per-wave unfolded x tile: 16 pixels x 288, f16, padded rows ----
  const int st = blockIdx.x * WAVES + wave;                 // 0..783
  const int b  = st / TPI;
  const int l0 = (st - b * TPI) * 16;
  _Float16* xw = &xu[wave * 16 * XSTR];
  {
    const int n  = lane & 15;                               // pixel in tile
    const int kb = (lane >> 4) * (CKK / 2);                 // k half: 0 / 144
    const int l  = l0 + n;
    const int oh = l / HW;
    const int ow = l - oh * HW;
    const float* xb = x + (size_t)b * NC * NPIX;
    for (int kk = 0; kk < CKK / 2; ++kk) {
      const int k  = kb + kk;                               // k = c*9 + di*3 + dj
      const int c  = k / 9;
      const int r  = k - c * 9;
      const int di = r / 3;
      const int dj = r - di * 3;
      const int h  = oh + di - 1;
      const int ww = ow + dj - 1;
      float v = 0.f;                                        // zero padding
      if ((unsigned)h < (unsigned)HW && (unsigned)ww < (unsigned)HW)
        v = xb[(c * HW + h) * HW + ww];
      xw[n * XSTR + k] = (_Float16)v;
    }
  }
  __syncthreads();   // publishes smu; xu is wave-private (DS in-order per wave)

  // ---- Preload this lane's 144 x-values into VGPRs (reused for all 64 o) ----
  // Lane n holds k=[kc*32 .. +16), lane 16+n holds k=[kc*32+16 .. +32) of pixel n:
  // together the B-operand column n covers all 32 k per WMMA chunk.
  const _Float16* xcol = xw + (lane & 15) * XSTR + (lane >> 4) * 16;
  v4u xr0[9], xr1[9];
#pragma unroll
  for (int kc = 0; kc < 9; ++kc) {
    xr0[kc] = *(const v4u*)(xcol + kc * 32);
    xr1[kc] = *(const v4u*)(xcol + kc * 32 + 8);
  }

  v16h ones;
#pragma unroll
  for (int i = 0; i < 16; ++i) ones[i] = (_Float16)1.0f;

  const float mu = smu;
  const _Float16* wbase = wh + (lane >> 4) * 16;            // k-half select
  float* outp = out + ((size_t)b * NO) * NPIX + l0 + (lane & 15);

  // Two output channels per iteration: independent WMMA accumulate chains
  // interleave, hiding the 5-slot WMMA->WMMA D->C hazard (no v_nop stalls).
  for (int o = 0; o < NO; o += 2) {
    const _Float16* wrowA = wbase + o * CKK;
    const _Float16* wrowB = wrowA + CKK;
    v8f accA, accB;
#pragma unroll
    for (int i = 0; i < 8; ++i) { accA[i] = 0.f; accB[i] = 0.f; }
#pragma unroll
    for (int kc = 0; kc < 9; ++kc) {                        // 9*32 = 288 = K
      const v4u wa0 = *(const v4u*)(wrowA + kc * 32);       // LDS broadcast
      const v4u wa1 = *(const v4u*)(wrowA + kc * 32 + 8);
      const v4u wb0 = *(const v4u*)(wrowB + kc * 32);
      const v4u wb1 = *(const v4u*)(wrowB + kc * 32 + 8);
      const v8h ma0 = min8(xr0[kc], wa0);                   // v_pk_min_num_f16
      const v8h ma1 = min8(xr1[kc], wa1);
      const v8h mb0 = min8(xr0[kc], wb0);
      const v8h mb1 = min8(xr1[kc], wb1);
      const v16h mA = __builtin_shufflevector(ma0, ma1,
                        0,1,2,3,4,5,6,7,8,9,10,11,12,13,14,15);
      const v16h mB = __builtin_shufflevector(mb0, mb1,
                        0,1,2,3,4,5,6,7,8,9,10,11,12,13,14,15);
      // D = ones(16x32) x m(32x16) + C  => every D row = column sums of m
      accA = __builtin_amdgcn_wmma_f32_16x16x32_f16(
               false, ones, false, mA, (short)0, accA, false, false);
      accB = __builtin_amdgcn_wmma_f32_16x16x32_f16(
               false, ones, false, mB, (short)0, accB, false, false);
    }
    // All D rows are identical column sums. Lanes 0-15 (row M=0, N=lane) store
    // channel o; lanes 16-31 (row M=8, N=lane-16) store channel o+1: full-EXEC.
    const float v = mu * ((lane < 16) ? accA[0] : accB[0]);
    outp[(size_t)(o + (lane >> 4)) * NPIX] = v;
  }
}

extern "C" void kernel_launch(void* const* d_in, const int* in_sizes, int n_in,
                              void* d_out, int out_size, void* d_ws, size_t ws_size,
                              hipStream_t stream) {
  (void)in_sizes; (void)n_in; (void)d_ws; (void)ws_size; (void)out_size;
  const float* x = (const float*)d_in[0];   // (4,32,56,56) f32
  const float* w = (const float*)d_in[1];   // (64,32,3,3)  f32
  float* out = (float*)d_out;               // (4,64,56,56) f32
  // 784 16-pixel tiles total, 2 waves (tiles) per 64-thread workgroup
  minplus_conv2d_kernel<<<dim3((NB * TPI) / WAVES), dim3(THREADS), 0, stream>>>(x, w, out);
}
